// GatedLinearAttention_2207613190881
// MI455X (gfx1250) — matrix operand order repaired and checked by reading, not compiled, not run on hardware
//
#include <hip/hip_runtime.h>
#include <stdint.h>

// ---- problem constants (from reference) ----
#define B_    2
#define L_    2048
#define HID_  1024
#define H_    16
#define Dh    64
#define LR_   16
#define EPS_  1e-5f
#define GLN_  16.0f
#define SCALE_ 0.125f           // Dh^-0.5
#define M_    (B_ * L_)         // 4096 rows in all big GEMMs

// ---- CDNA5 WMMA vector types ----
typedef __attribute__((ext_vector_type(16))) __bf16        v16bf;
typedef __attribute__((ext_vector_type(8)))  float         v8f;
typedef __attribute__((ext_vector_type(4)))  unsigned int  u32x4;

struct U32x4Pair { u32x4 a, b; };   // 32B container -> bit_cast to v16bf

static __device__ __forceinline__ uint16_t f32_to_bf16_rne(float f) {
  uint32_t u = __builtin_bit_cast(uint32_t, f);
  u += 0x7FFFu + ((u >> 16) & 1u);  // round to nearest even
  return (uint16_t)(u >> 16);
}

// CDNA5 async memory->LDS copy (16B per lane), tracked by ASYNCcnt.
static __device__ __forceinline__ void async_b128_to_lds(uint16_t* lds_dst,
                                                         const uint16_t* gsrc) {
  uint32_t lds_off = (uint32_t)(uintptr_t)lds_dst;   // flat LDS aperture: addr[31:0]
  uint64_t gaddr   = (uint64_t)(uintptr_t)gsrc;
  asm volatile("global_load_async_to_lds_b128 %0, %1, off"
               :
               : "v"(lds_off), "v"(gaddr)
               : "memory");
}

static __device__ __forceinline__ void wait_async_lds() {
  asm volatile("s_wait_asynccnt 0" ::: "memory");
}

// ---------------------------------------------------------------------------
// f32 -> bf16 elementwise conversion
// ---------------------------------------------------------------------------
__global__ void cvt_f32_bf16_kernel(const float* __restrict__ in,
                                    uint16_t* __restrict__ out, int n) {
  int i = blockIdx.x * blockDim.x + threadIdx.x;
  if (i < n) out[i] = f32_to_bf16_rne(in[i]);
}

// ---------------------------------------------------------------------------
// f32 [K,N] -> bf16 transposed [N,K] (one-time weight prep; amortized over
// all GEMM tile passes so B-tiles load k-contiguous like A-tiles)
// ---------------------------------------------------------------------------
__global__ void cvt_transpose_f32_bf16_kernel(const float* __restrict__ in,  // [1024,1024]
                                              uint16_t* __restrict__ out) { // [1024,1024]^T
  int gid = blockIdx.x * blockDim.x + threadIdx.x;
  if (gid >= HID_ * HID_) return;
  int k = gid & (HID_ - 1);   // k fastest -> coalesced writes
  int n = gid >> 10;
  out[gid] = f32_to_bf16_rne(in[(size_t)k * HID_ + n]);
}

// ---------------------------------------------------------------------------
// WMMA bf16 GEMM: C[M,N](f32) = A[M,K](bf16,row-major) @ Bt[N,K](bf16)^T
// Block tile 128x128, 256 threads (8 waves in 4x2 grid), wave tile 32x64
// (2x4 subtiles of 16x16), K-step 32 via v_wmma_f32_16x16x32_bf16.
// Tile fill uses global_load_async_to_lds_b128 (ASYNCcnt) - no VGPR staging.
// ---------------------------------------------------------------------------
#define BLK_M 128
#define BLK_N 128
#define BLK_K 32
#define LDSR  40   // padded LDS row stride in bf16 elems (80B = 5*16B, keeps b128 alignment)

__global__ __launch_bounds__(256)
void wmma_gemm_bf16(const uint16_t* __restrict__ A,    // [M,K] row-major
                    const uint16_t* __restrict__ Bt,   // [N,K] row-major (pre-transposed W)
                    float* __restrict__ C, int M, int N, int K) {
  __shared__ __align__(16) uint16_t sA[BLK_M * LDSR];   // A tile: [row][k]
  __shared__ __align__(16) uint16_t sB[BLK_N * LDSR];   // B tile: [col][k]

  const int t     = threadIdx.x;
  const int lane  = t & 31;
  const int wid   = t >> 5;        // 0..7
  const int wm    = wid & 3;       // 4 waves along M
  const int wn    = wid >> 2;      // 2 waves along N
  const int m0    = blockIdx.x * BLK_M;
  const int n0    = blockIdx.y * BLK_N;
  const int l15   = lane & 15;
  const int lhalf = (lane >> 4) & 1;

  v8f acc[2][4] = {};

  for (int k0 = 0; k0 < K; k0 += BLK_K) {
    __syncthreads();   // previous iteration's consumers done before overwrite
    // --- A & B tiles: 128 rows x 32 bf16 each; 2+2 async b128 per thread ---
#pragma unroll
    for (int i = 0; i < 2; i++) {
      int idx = t + i * 256;                 // 0..511
      int row = idx >> 2;
      int seg = idx & 3;                     // 4 x 8-elem segments per row
      async_b128_to_lds(sA + row * LDSR + seg * 8,
                        A + (size_t)(m0 + row) * K + k0 + seg * 8);
      async_b128_to_lds(sB + row * LDSR + seg * 8,
                        Bt + (size_t)(n0 + row) * K + k0 + seg * 8);
    }
    wait_async_lds();
    __syncthreads();

    // --- fragments per ISA 16-bit A/B lane layout ---
    // A 16x32: lanes 0-15 -> K {0..7,16..23}; lanes 16-31 -> K {8..15,24..31}
    v16bf afr[2];
    const int klo = lhalf * 8;
#pragma unroll
    for (int i = 0; i < 2; i++) {
      int row = wm * 32 + i * 16 + l15;
      U32x4Pair p;
      p.a = *(const u32x4*)(sA + row * LDSR + klo);
      p.b = *(const u32x4*)(sA + row * LDSR + 16 + klo);
      afr[i] = __builtin_bit_cast(v16bf, p);
    }
    // B 32x16: lanes 0-15 -> K {0..7,16..23} of column l15; lanes 16-31 -> K {8..15,24..31}
    v16bf bfr[4];
    const int kb = lhalf * 8;
#pragma unroll
    for (int j = 0; j < 4; j++) {
      int col = wn * 64 + j * 16 + l15;
      U32x4Pair p;
      p.a = *(const u32x4*)(sB + col * LDSR + kb);
      p.b = *(const u32x4*)(sB + col * LDSR + kb + 16);
      bfr[j] = __builtin_bit_cast(v16bf, p);
    }
#pragma unroll
    for (int i = 0; i < 2; i++)
#pragma unroll
      for (int j = 0; j < 4; j++)
        acc[i][j] = __builtin_amdgcn_wmma_f32_16x16x32_bf16(
            false, afr[i], false, bfr[j], (short)0, acc[i][j], false, false);
  }

  // --- store: C/D layout VGPR r -> M = r + 8*lhalf, N = l15 ---
#pragma unroll
  for (int i = 0; i < 2; i++)
#pragma unroll
    for (int j = 0; j < 4; j++) {
      int mbase = m0 + wm * 32 + i * 16 + lhalf * 8;
      int nc    = n0 + wn * 64 + j * 16 + l15;
#pragma unroll
      for (int r = 0; r < 8; r++)
        C[(size_t)(mbase + r) * N + nc] = acc[i][j][r];
    }
}

// ---------------------------------------------------------------------------
// Gate low-rank: xg1[M,16] = x[M,1024] @ Wg1[1024,16]
// ---------------------------------------------------------------------------
__global__ void gate_lowrank_kernel(const float* __restrict__ x,
                                    const float* __restrict__ Wg1,
                                    float* __restrict__ xg1) {
  int gid = blockIdx.x * blockDim.x + threadIdx.x;
  if (gid >= M_ * LR_) return;
  int m = gid >> 4, r = gid & 15;
  const float* xr = x + (size_t)m * HID_;
  float acc = 0.f;
  for (int k = 0; k < HID_; k++) acc = fmaf(xr[k], Wg1[k * LR_ + r], acc);
  xg1[gid] = acc;
}

// ---------------------------------------------------------------------------
// gk[M,1024] = logsigmoid(xg1 @ Wg2 + bg2) / GLN
// ---------------------------------------------------------------------------
__global__ void gate_expand_kernel(const float* __restrict__ xg1,
                                   const float* __restrict__ Wg2,
                                   const float* __restrict__ bg2,
                                   float* __restrict__ gk) {
  int gid = blockIdx.x * blockDim.x + threadIdx.x;
  if (gid >= M_ * HID_) return;
  int m = gid / HID_, n = gid % HID_;
  float acc = bg2[n];
#pragma unroll
  for (int r = 0; r < LR_; r++)
    acc = fmaf(xg1[m * LR_ + r], Wg2[r * HID_ + n], acc);
  float ls = fminf(acc, 0.f) - log1pf(__expf(-fabsf(acc)));   // stable logsigmoid
  gk[gid] = ls * (1.0f / GLN_);
}

// ---------------------------------------------------------------------------
// Sequential GLA scan. One block per (b,h); 64 threads; thread j owns state
// column S[:, j] so o_j = sum_d S[d][j]*q[d] is fully thread-local.
// ---------------------------------------------------------------------------
__global__ __launch_bounds__(64)
void gla_scan_kernel(const float* __restrict__ q, const float* __restrict__ k,
                     const float* __restrict__ v, const float* __restrict__ gk,
                     float* __restrict__ o, float* __restrict__ state) {
  const int bh = blockIdx.x;
  const int b  = bh / H_;
  const int h  = bh % H_;
  const int j  = threadIdx.x;

  __shared__ __align__(16) float qs[Dh];
  __shared__ __align__(16) float ks[Dh];
  __shared__ __align__(16) float egs[Dh];

  float S[Dh];
#pragma unroll
  for (int d = 0; d < Dh; d++) S[d] = 0.f;

  const size_t base   = ((size_t)b * L_ * H_ + h) * Dh;   // (b, t=0, h, 0)
  const size_t stride = (size_t)H_ * Dh;                  // 1024 per timestep

  for (int t = 0; t < L_; t++) {
    const size_t off = base + (size_t)t * stride;
    qs[j]  = q[off + j];
    ks[j]  = k[off + j];
    egs[j] = __expf(gk[off + j]);
    const float vj = v[off + j];
    __syncthreads();

    float oj = 0.f;
#pragma unroll
    for (int d = 0; d < Dh; d += 4) {
      float4 qd = *(const float4*)(qs + d);
      float4 kd = *(const float4*)(ks + d);
      float4 ed = *(const float4*)(egs + d);
      float s;
      s = fmaf(ed.x, S[d + 0], kd.x * vj); S[d + 0] = s; oj = fmaf(s, qd.x, oj);
      s = fmaf(ed.y, S[d + 1], kd.y * vj); S[d + 1] = s; oj = fmaf(s, qd.y, oj);
      s = fmaf(ed.z, S[d + 2], kd.z * vj); S[d + 2] = s; oj = fmaf(s, qd.z, oj);
      s = fmaf(ed.w, S[d + 3], kd.w * vj); S[d + 3] = s; oj = fmaf(s, qd.w, oj);
    }
    o[off + j] = oj * SCALE_;
    __syncthreads();   // protect LDS before next timestep overwrites it
  }

  // last_state: (B,1,H,D,D) -> state[((b*H+h)*D + d)*D + j]
#pragma unroll
  for (int d = 0; d < Dh; d++)
    state[((size_t)(b * H_ + h) * Dh + d) * Dh + j] = S[d];
}

// ---------------------------------------------------------------------------
// RMSNorm over head dim + affine weight + swish output gate; emit bf16 rows
// for the final Wo GEMM. One thread per (b,l,h) head (contiguous 64 floats).
// ---------------------------------------------------------------------------
__global__ void rms_swish_kernel(const float* __restrict__ o_att,
                                 const float* __restrict__ g,
                                 const float* __restrict__ gw,
                                 uint16_t* __restrict__ obf) {
  int i = blockIdx.x * blockDim.x + threadIdx.x;
  if (i >= M_ * H_) return;
  const float* op = o_att + (size_t)i * Dh;
  const float* gp = g     + (size_t)i * Dh;   // (b*L+l)*1024 + h*64 == i*64
  float ss = 0.f;
  for (int d = 0; d < Dh; d++) { float x = op[d]; ss = fmaf(x, x, ss); }
  const float r = rsqrtf(ss * (1.0f / Dh) + EPS_);
  uint16_t* ob = obf + (size_t)i * Dh;
  for (int d = 0; d < Dh; d++) {
    float gv = gp[d];
    float sw = gv / (1.f + __expf(-gv));      // g * sigmoid(g)
    ob[d] = f32_to_bf16_rne(op[d] * r * gw[d] * sw);
  }
}

// ---------------------------------------------------------------------------
// launcher
// ---------------------------------------------------------------------------
extern "C" void kernel_launch(void* const* d_in, const int* in_sizes, int n_in,
                              void* d_out, int out_size, void* d_ws, size_t ws_size,
                              hipStream_t stream) {
  const float* x   = (const float*)d_in[0];
  const float* Wq  = (const float*)d_in[1];
  const float* Wk  = (const float*)d_in[2];
  const float* Wv  = (const float*)d_in[3];
  const float* Wg1 = (const float*)d_in[4];
  const float* Wg2 = (const float*)d_in[5];
  const float* bg2 = (const float*)d_in[6];
  const float* Wg  = (const float*)d_in[7];
  const float* gnw = (const float*)d_in[8];
  const float* Wo  = (const float*)d_in[9];
  float* out = (float*)d_out;

  // ---- workspace partition ----
  char* ws = (char*)d_ws;
  size_t off = 0;
  uint16_t* xb  = (uint16_t*)(ws + off); off += (size_t)M_ * HID_ * 2;
  uint16_t* Wqt = (uint16_t*)(ws + off); off += (size_t)HID_ * HID_ * 2;
  uint16_t* Wkt = (uint16_t*)(ws + off); off += (size_t)HID_ * HID_ * 2;
  uint16_t* Wvt = (uint16_t*)(ws + off); off += (size_t)HID_ * HID_ * 2;
  uint16_t* Wgt = (uint16_t*)(ws + off); off += (size_t)HID_ * HID_ * 2;
  uint16_t* Wot = (uint16_t*)(ws + off); off += (size_t)HID_ * HID_ * 2;
  float* qf   = (float*)(ws + off); off += (size_t)M_ * HID_ * 4;
  float* kf   = (float*)(ws + off); off += (size_t)M_ * HID_ * 4;
  float* vf   = (float*)(ws + off); off += (size_t)M_ * HID_ * 4;
  float* gf   = (float*)(ws + off); off += (size_t)M_ * HID_ * 4;
  float* gkf  = (float*)(ws + off); off += (size_t)M_ * HID_ * 4;
  float* xg1f = (float*)(ws + off); off += (size_t)M_ * LR_ * 4;
  float* oatt = (float*)(ws + off); off += (size_t)M_ * HID_ * 4;
  uint16_t* obf = (uint16_t*)(ws + off); off += (size_t)M_ * HID_ * 2;
  (void)ws_size; (void)in_sizes; (void)n_in; (void)out_size;

  const int nX = M_ * HID_;
  const int nW = HID_ * HID_;

  // 1) bf16 conversions (weights transposed once -> k-contiguous B tiles)
  cvt_f32_bf16_kernel<<<(nX + 255) / 256, 256, 0, stream>>>(x, xb, nX);
  cvt_transpose_f32_bf16_kernel<<<(nW + 255) / 256, 256, 0, stream>>>(Wq, Wqt);
  cvt_transpose_f32_bf16_kernel<<<(nW + 255) / 256, 256, 0, stream>>>(Wk, Wkt);
  cvt_transpose_f32_bf16_kernel<<<(nW + 255) / 256, 256, 0, stream>>>(Wv, Wvt);
  cvt_transpose_f32_bf16_kernel<<<(nW + 255) / 256, 256, 0, stream>>>(Wg, Wgt);
  cvt_transpose_f32_bf16_kernel<<<(nW + 255) / 256, 256, 0, stream>>>(Wo, Wot);

  // 2) q/k/v/g projections via WMMA
  dim3 ggrid(M_ / BLK_M, HID_ / BLK_N);
  wmma_gemm_bf16<<<ggrid, 256, 0, stream>>>(xb, Wqt, qf, M_, HID_, HID_);
  wmma_gemm_bf16<<<ggrid, 256, 0, stream>>>(xb, Wkt, kf, M_, HID_, HID_);
  wmma_gemm_bf16<<<ggrid, 256, 0, stream>>>(xb, Wvt, vf, M_, HID_, HID_);
  wmma_gemm_bf16<<<ggrid, 256, 0, stream>>>(xb, Wgt, gf, M_, HID_, HID_);

  // 3) gate path
  gate_lowrank_kernel<<<(M_ * LR_ + 255) / 256, 256, 0, stream>>>(x, Wg1, xg1f);
  gate_expand_kernel<<<(nX + 255) / 256, 256, 0, stream>>>(xg1f, Wg2, bg2, gkf);

  // 4) sequential scan; writes o_att and the final state region of d_out
  gla_scan_kernel<<<B_ * H_, 64, 0, stream>>>(qf, kf, vf, gkf, oatt,
                                              out + (size_t)M_ * HID_);

  // 5) RMSNorm + swish gate -> bf16
  rms_swish_kernel<<<(M_ * H_ + 255) / 256, 256, 0, stream>>>(oatt, gf, gnw, obf);

  // 6) final projection into d_out
  wmma_gemm_bf16<<<ggrid, 256, 0, stream>>>(obf, Wot, out, M_, HID_, HID_);
}
